// CatanGNN_66692252172455
// MI455X (gfx1250) — matrix operand order repaired and checked by reading, not compiled, not run on hardware
//
#include <hip/hip_runtime.h>

#define NNODES 100000
#define NEDGES 300000
#define HIDF   128

typedef __attribute__((ext_vector_type(16))) _Float16 v16h;
typedef __attribute__((ext_vector_type(8)))  _Float16 v8h;
typedef __attribute__((ext_vector_type(8)))  float    v8f;

union V16H { v16h v; v8h h2[2]; _Float16 e[16]; };

// ---------------------------------------------------------------------------
// In-projection: Y[N x 128] = X[N x 16] @ W[16 x 128] + b   (K=16, trivial)
// ---------------------------------------------------------------------------
__global__ __launch_bounds__(256)
void gnn_inproj(const float* __restrict__ X, const float* __restrict__ W,
                const float* __restrict__ b, float* __restrict__ Y, int Nn)
{
    int i = blockIdx.x * 256 + threadIdx.x;
    if (i >= Nn * HIDF) return;
    int row = i >> 7, col = i & 127;
    const float* xr = X + (size_t)row * 16;
    float s = b[col];
#pragma unroll
    for (int k = 0; k < 16; ++k) s = fmaf(xr[k], W[k * HIDF + col], s);
    Y[i] = s;
}

// ---------------------------------------------------------------------------
// WMMA GEMM: Y[M x 128] = X[M x 128] @ W[128 x 128] (f32 IO, f16 WMMA, f32 acc)
// Block = 128 rows x 128 cols; 8 waves, each wave a 16-row strip of 8 tiles.
// W staged to LDS as f16 transposed [n][k], stride 136 halves (pad => no bank
// conflicts for the 16B ds_load chunks of B fragments). All 16 A-strip loads
// are hoisted into distinct registers so global latency is paid once, then
// the 32-WMMA pipeline runs purely off registers + LDS.
// ---------------------------------------------------------------------------
__global__ __launch_bounds__(256)
void gnn_gemm_n128(const float* __restrict__ X, const float* __restrict__ W,
                   float* __restrict__ Y, int M)
{
    __shared__ __align__(16) _Float16 Wl[128 * 136];

    for (int idx = threadIdx.x; idx < 4096; idx += 256) {
        int k = idx >> 5;            // 0..127 (K row of W)
        int n = (idx & 31) << 2;     // 0..124 (N col, float4)
        float4 w = *(const float4*)(W + (size_t)k * 128 + n);
        Wl[(n + 0) * 136 + k] = (_Float16)w.x;
        Wl[(n + 1) * 136 + k] = (_Float16)w.y;
        Wl[(n + 2) * 136 + k] = (_Float16)w.z;
        Wl[(n + 3) * 136 + k] = (_Float16)w.w;
    }
    __syncthreads();

    const int lane = threadIdx.x & 31;
    const int wave = threadIdx.x >> 5;
    const int half = lane >> 4;      // ISA A/B lane-half (K split)
    const int l15  = lane & 15;
    const int rowBase = blockIdx.x * 128 + wave * 16;
    int m  = rowBase + l15;
    int mc = m < M ? m : (M - 1);    // clamp tail rows (stores are guarded)

    // --- hoist all A loads (4 K-chunks x 4 float4) -------------------------
    float4 ar[16];
    const float* xbase = X + (size_t)mc * 128 + half * 8;
#pragma unroll
    for (int kc = 0; kc < 4; ++kc) {
        const float* xr = xbase + kc * 32;
        ar[kc * 4 + 0] = *(const float4*)(xr);
        ar[kc * 4 + 1] = *(const float4*)(xr + 4);
        ar[kc * 4 + 2] = *(const float4*)(xr + 16);
        ar[kc * 4 + 3] = *(const float4*)(xr + 20);
    }

    V16H Af[4];
#pragma unroll
    for (int kc = 0; kc < 4; ++kc) {
#pragma unroll
        for (int j = 0; j < 4; ++j) {
            float4 a = ar[kc * 4 + j];
            Af[kc].e[j * 4 + 0] = (_Float16)a.x;
            Af[kc].e[j * 4 + 1] = (_Float16)a.y;
            Af[kc].e[j * 4 + 2] = (_Float16)a.z;
            Af[kc].e[j * 4 + 3] = (_Float16)a.w;
        }
    }

    v8f acc[8] = {};

#pragma unroll
    for (int kc = 0; kc < 4; ++kc) {
#pragma unroll
        for (int t = 0; t < 8; ++t) {
            // B fragment (32x16): lane column n = t*16+l15, K = 8h..8h+7, 16+8h..16+8h+7.
            const _Float16* wp = &Wl[(t * 16 + l15) * 136 + kc * 32 + half * 8];
            V16H B;
            B.h2[0] = *(const v8h*)(wp);
            B.h2[1] = *(const v8h*)(wp + 16);
            acc[t] = __builtin_amdgcn_wmma_f32_16x16x32_f16(
                false, Af[kc].v, false, B.v, (short)0, acc[t], false, false);
        }
    }

#pragma unroll
    for (int t = 0; t < 8; ++t) {
#pragma unroll
        for (int r = 0; r < 8; ++r) {
            int row = rowBase + half * 8 + r;   // C/D layout: VGPR r, lane half
            if (row < M) Y[(size_t)row * 128 + t * 16 + l15] = acc[t][r];
        }
    }
}

// ---------------------------------------------------------------------------
// Edge kernels: one wave32 per edge; lane*4 => exactly one 512B feature row.
// ---------------------------------------------------------------------------
__device__ __forceinline__ float gnn_lrelu(float x) { return x > 0.0f ? x : 0.2f * x; }

__device__ __forceinline__ void gnn_atomic_max_f(float* addr, float val)
{
    unsigned int* ua = (unsigned int*)addr;
    unsigned int old = *ua;
    while (__uint_as_float(old) < val) {
        unsigned int assumed = old;
        old = atomicCAS(ua, assumed, __float_as_uint(val));
        if (old == assumed) break;
    }
}

__global__ __launch_bounds__(256)
void gnn_edge_logits(const int* __restrict__ ei, const float* __restrict__ GL,
                     const float* __restrict__ GR, const float* __restrict__ att,
                     float* __restrict__ ebuf, float* __restrict__ mx, int E_)
{
    int wid = blockIdx.x * 8 + (threadIdx.x >> 5);
    if (wid >= E_) return;
    int lane = threadIdx.x & 31;
    int src = ei[wid];
    int dst = ei[E_ + wid];
    float4 gl = *(const float4*)(GL + (size_t)src * 128 + lane * 4);
    float4 gr = *(const float4*)(GR + (size_t)dst * 128 + lane * 4);
    float4 at = *(const float4*)(att + lane * 4);
    float s = gnn_lrelu(gl.x + gr.x) * at.x + gnn_lrelu(gl.y + gr.y) * at.y
            + gnn_lrelu(gl.z + gr.z) * at.z + gnn_lrelu(gl.w + gr.w) * at.w;
    // per-head reduction over each 16-lane half (head = lane>>4)
    s += __shfl_xor(s, 8, 32);
    s += __shfl_xor(s, 4, 32);
    s += __shfl_xor(s, 2, 32);
    s += __shfl_xor(s, 1, 32);
    if ((lane & 15) == 0) {
        int hh = lane >> 4;
        ebuf[(size_t)wid * 2 + hh] = s;
        gnn_atomic_max_f(mx + (size_t)dst * 2 + hh, s);
    }
}

__global__ __launch_bounds__(256)
void gnn_edge_scatter(const int* __restrict__ ei, const float* __restrict__ GL,
                      const float* __restrict__ ebuf, const float* __restrict__ mx,
                      float* __restrict__ denom, float* __restrict__ num, int E_)
{
    int wid = blockIdx.x * 8 + (threadIdx.x >> 5);
    if (wid >= E_) return;
    int lane = threadIdx.x & 31;
    int hh = lane >> 4;
    int src = ei[wid];
    int dst = ei[E_ + wid];
    float e = ebuf[(size_t)wid * 2 + hh];
    float m = mx[(size_t)dst * 2 + hh];
    float p = __expf(e - m);
    if ((lane & 15) == 0) atomicAdd(denom + (size_t)dst * 2 + hh, p);
    float4 gl = *(const float4*)(GL + (size_t)src * 128 + lane * 4);
    float* np = num + (size_t)dst * 128 + lane * 4;
    atomicAdd(np + 0, p * gl.x);
    atomicAdd(np + 1, p * gl.y);
    atomicAdd(np + 2, p * gl.z);
    atomicAdd(np + 3, p * gl.w);
}

// ---------------------------------------------------------------------------
// Elementwise helpers
// ---------------------------------------------------------------------------
__global__ __launch_bounds__(256)
void gnn_fill_f32(float* __restrict__ p, float v, int n)
{
    int i = blockIdx.x * 256 + threadIdx.x;
    if (i < n) p[i] = v;
}

__global__ __launch_bounds__(256)
void gnn_init_accum(float* __restrict__ A, const float* __restrict__ b0,
                    const float* __restrict__ b1, const float* __restrict__ b2, int n)
{
    int i = blockIdx.x * 256 + threadIdx.x;
    if (i >= n) return;
    int c = i & 127;
    A[i] = b0[c] + b1[c] + b2[c];
}

__global__ __launch_bounds__(256)
void gnn_finalize_rel(float* __restrict__ A, const float* __restrict__ num,
                      const float* __restrict__ denom, int n)
{
    int i = blockIdx.x * 256 + threadIdx.x;
    if (i >= n) return;
    int node = i >> 7;
    int hh = (i >> 6) & 1;
    A[i] += num[i] / (denom[(size_t)node * 2 + hh] + 1e-16f);
}

__global__ __launch_bounds__(256)
void gnn_relu_copy(const float* __restrict__ A, float* __restrict__ H, int n)
{
    int i = blockIdx.x * 256 + threadIdx.x;
    if (i >= n) return;
    float v = A[i];
    H[i] = v > 0.0f ? v : 0.0f;
}

// ---------------------------------------------------------------------------
// Orchestration
// ---------------------------------------------------------------------------
extern "C" void kernel_launch(void* const* d_in, const int* in_sizes, int n_in,
                              void* d_out, int out_size, void* d_ws, size_t ws_size,
                              hipStream_t stream)
{
    (void)in_sizes; (void)n_in; (void)out_size; (void)ws_size;
    const int N = NNODES, E = NEDGES;
    const size_t NH = (size_t)N * HIDF;

    const float* x[3]   = {(const float*)d_in[0], (const float*)d_in[1], (const float*)d_in[2]};
    const int*   ei[9];
    for (int r = 0; r < 9; ++r) ei[r] = (const int*)d_in[3 + r];
    const float* Win[3] = {(const float*)d_in[12], (const float*)d_in[14], (const float*)d_in[16]};
    const float* bin[3] = {(const float*)d_in[13], (const float*)d_in[15], (const float*)d_in[17]};

    float* ws    = (float*)d_ws;
    float* h[3]  = {ws, ws + NH, ws + 2 * NH};     // current node features
    float* GL    = ws + 3 * NH;                    // src projection
    float* GR    = ws + 4 * NH;                    // dst projection
    float* num   = ws + 5 * NH;                    // softmax numerator scatter
    float* ebuf  = ws + 6 * NH;                    // per-edge logits [E,2]
    float* mx    = ebuf + (size_t)2 * E;           // per-(dst,head) max  [N,2]
    float* denom = mx + (size_t)2 * N;             // per-(dst,head) sum  [N,2]

    float* outp = (float*)d_out;

    const int eb = (N * HIDF + 255) / 256;         // elementwise blocks
    const int gemmBlocks = (N + 127) / 128;
    const int edgeBlocks = (E + 7) / 8;

    for (int t = 0; t < 3; ++t)
        gnn_inproj<<<eb, 256, 0, stream>>>(x[t], Win[t], bin[t], h[t], N);

    // relation r: src type, dst type  (0=tile,1=vertex,2=edge)
    static const int srcT[9] = {0, 0, 1, 0, 2, 1, 2, 1, 2};
    static const int dstT[9] = {0, 1, 0, 2, 0, 2, 1, 1, 2};
    static const int relOfDst[3][3] = {{0, 2, 4}, {1, 6, 7}, {3, 5, 8}};

    for (int layer = 0; layer < 2; ++layer) {
        const float* Wlw = (const float*)d_in[18 + layer * 4];
        const float* Wrw = (const float*)d_in[19 + layer * 4];
        const float* att = (const float*)d_in[20 + layer * 4];
        const float* bc  = (const float*)d_in[21 + layer * 4];

        // Accumulators live directly in d_out for both layers.
        float* A[3] = {outp, outp + NH, outp + 2 * NH};

        for (int t = 0; t < 3; ++t)
            gnn_init_accum<<<eb, 256, 0, stream>>>(A[t],
                bc + relOfDst[t][0] * HIDF, bc + relOfDst[t][1] * HIDF,
                bc + relOfDst[t][2] * HIDF, N * HIDF);

        for (int r = 0; r < 9; ++r) {
            gnn_gemm_n128<<<gemmBlocks, 256, 0, stream>>>(
                h[srcT[r]], Wlw + (size_t)r * HIDF * HIDF, GL, N);
            gnn_gemm_n128<<<gemmBlocks, 256, 0, stream>>>(
                h[dstT[r]], Wrw + (size_t)r * HIDF * HIDF, GR, N);

            gnn_fill_f32<<<(2 * N + 255) / 256, 256, 0, stream>>>(mx, -__builtin_inff(), 2 * N);
            hipMemsetAsync(denom, 0, (size_t)2 * N * sizeof(float), stream);
            hipMemsetAsync(num, 0, NH * sizeof(float), stream);

            gnn_edge_logits<<<edgeBlocks, 256, 0, stream>>>(
                ei[r], GL, GR, att + r * HIDF, ebuf, mx, E);
            gnn_edge_scatter<<<edgeBlocks, 256, 0, stream>>>(
                ei[r], GL, ebuf, mx, denom, num, E);
            gnn_finalize_rel<<<eb, 256, 0, stream>>>(A[dstT[r]], num, denom, N * HIDF);
        }

        if (layer == 0)
            for (int t = 0; t < 3; ++t)
                gnn_relu_copy<<<eb, 256, 0, stream>>>(A[t], h[t], N * HIDF);
    }
}
